// AttentionHead_2791728742648
// MI455X (gfx1250) — matrix operand order repaired and checked by reading, not compiled, not run on hardware
//
#include <hip/hip_runtime.h>

typedef _Float16 v16h __attribute__((ext_vector_type(16)));
typedef _Float16 v8h  __attribute__((ext_vector_type(8)));
typedef float    v8f  __attribute__((ext_vector_type(8)));

#define BATCH 4
#define SEQ   2048
#define DIM   1024
#define HD    64
#define ROWS  (BATCH*SEQ)   // 8192

// 1/sqrt(64) * log2(e): softmax computed in base-2 (v_exp_f32 is base-2 native)
#define SCALE_LOG2 (0.125f * 1.44269504088896340736f)

static __device__ __forceinline__ v8f wmma_f16(v16h a, v16h b, v8f c) {
    return __builtin_amdgcn_wmma_f32_16x16x32_f16(false, a, false, b, (short)0, c, false, false);
}

// ---------------------------------------------------------------------------
// Kernel 0: transpose + f16-convert the three weight matrices.
// Wt layout: [which][n (0..63)][k (0..1023)] f16, so WMMA B-fragments are
// contiguous 32B per lane.
// ---------------------------------------------------------------------------
__global__ __launch_bounds__(256) void wt_kernel(
    const float* __restrict__ Wq, const float* __restrict__ Wk,
    const float* __restrict__ Wv, _Float16* __restrict__ Wt)
{
    int idx   = blockIdx.x * 256 + threadIdx.x;   // 0 .. 196607
    int which = idx >> 16;
    int rem   = idx & 65535;
    int n     = rem >> 10;
    int k     = rem & 1023;
    const float* W = (which == 0) ? Wq : (which == 1) ? Wk : Wv;
    Wt[idx] = (_Float16)W[k * HD + n];
}

// ---------------------------------------------------------------------------
// Kernel 1: fused QKV projection. GEMM [8192 x 1024] @ [1024 x 64] x3.
// One block per 16-row M tile; 12 waves = {q,k,v} x 4 head-tiles.
// x tile staged in LDS as f16 (padded stride: 2064B = 516 dwords, 516%64=4,
// so the 16 A-fragment rows hit distinct banks).
// Outputs: q,k natural f16 [8192][64]; v transposed f16 [B][64][SEQ].
// ---------------------------------------------------------------------------
#define XPAD 8
__global__ __launch_bounds__(384) void qkv_kernel(
    const float* __restrict__ x, const _Float16* __restrict__ Wt,
    const float* __restrict__ bq, const float* __restrict__ bk,
    const float* __restrict__ bv,
    _Float16* __restrict__ qf, _Float16* __restrict__ kf,
    _Float16* __restrict__ vtf)
{
    __shared__ _Float16 xs[16][DIM + XPAD];

    const int mtile = blockIdx.x;        // 0..511
    const int tid   = threadIdx.x;

    // Cooperative load + f32->f16 convert of the 16x1024 x tile (float4 reads).
    for (int i4 = tid; i4 < 4096; i4 += 384) {
        int row = i4 >> 8;
        int c4  = (i4 & 255) << 2;
        const float4 v = *(const float4*)&x[(mtile * 16 + row) * DIM + c4];
        xs[row][c4 + 0] = (_Float16)v.x;
        xs[row][c4 + 1] = (_Float16)v.y;
        xs[row][c4 + 2] = (_Float16)v.z;
        xs[row][c4 + 3] = (_Float16)v.w;
    }
    __syncthreads();

    const int wave = tid >> 5;
    const int lane = tid & 31;
    const int lr   = lane & 15;          // A: row / B,C: column
    const int lh   = lane >> 4;          // half-wave select

    const int which = wave >> 2;         // 0=q 1=k 2=v
    const int ntile = wave & 3;
    const int n     = ntile * 16 + lr;   // output head index for this lane
    const _Float16* Wrow = Wt + which * (HD * DIM) + n * DIM;

    v8f c = {};
    #pragma unroll 4
    for (int kc = 0; kc < 32; ++kc) {
        // A fragment: lane holds row lr, K = lh*16..+15 (contiguous)
        int ca = kc * 32 + lh * 16;
        v8h lo = *(const v8h*)&xs[lr][ca];
        v8h hi = *(const v8h*)&xs[lr][ca + 8];
        v16h a;
        #pragma unroll
        for (int i = 0; i < 8; ++i) { a[i] = lo[i]; a[i + 8] = hi[i]; }
        // B fragment: lane holds column n, K = lh*16..+15 (contiguous 32B)
        v16h b = *(const v16h*)&Wrow[kc * 32 + lh * 16];
        c = wmma_f16(a, b, c);
    }

    const float* bias = (which == 0) ? bq : (which == 1) ? bk : bv;
    const float  bn   = bias[n];
    const int    m0   = mtile * 16;

    if (which < 2) {
        _Float16* dst = (which == 0) ? qf : kf;
        #pragma unroll
        for (int r = 0; r < 8; ++r) {
            int m = m0 + r + 8 * lh;                 // C row mapping
            dst[m * HD + n] = (_Float16)(c[r] + bn);
        }
    } else {
        #pragma unroll
        for (int r = 0; r < 8; ++r) {
            int m  = m0 + r + 8 * lh;
            int bb = m >> 11;                        // batch
            int ss = m & 2047;                       // seq pos
            vtf[bb * (HD * SEQ) + n * SEQ + ss] = (_Float16)(c[r] + bn);
        }
    }
}

// ---------------------------------------------------------------------------
// Kernel 2: flash-style causal attention, one single-wave workgroup per
// 16-query tile (512 workgroups scatter across all WGPs). K/V chunk tiles are
// double-buffered in LDS with GLOBAL_LOAD_ASYNC_TO_LDS_B128 (ASYNCcnt),
// so chunk i+1 streams L2->LDS while chunk i runs WMMA + softmax.
// ---------------------------------------------------------------------------
#define PSTR  40   // f16 stride for p transpose buffer (80B, 16B-aligned)
#define KSTR  72   // f16 row stride for K tile (144B: bank-conflict-free)
#define VSTR  40   // f16 row stride for V tile (80B: bank-conflict-free)

// Issue 16 async 16B-per-lane copies: K tile (32 rows x 128B, contiguous in
// memory) and V tile (64 rows x 64B, row stride 4096B) into padded LDS.
static __device__ __forceinline__ void async_copy_chunk(
    const _Float16* kb, const _Float16* vb, int key0,
    _Float16* kdst, _Float16* vdst, int lane)
{
    const char* kg = (const char*)(kb + key0 * HD);   // 4KB contiguous
    #pragma unroll
    for (int i = 0; i < 8; ++i) {
        int seg = i * 32 + lane;                      // 256 x 16B segments
        unsigned int       ld = (unsigned int)(size_t)((char*)kdst + (seg >> 3) * (KSTR * 2) + (seg & 7) * 16);
        unsigned long long ga = (unsigned long long)(size_t)(kg + seg * 16);
        asm volatile("global_load_async_to_lds_b128 %0, %1, off"
                     :: "v"(ld), "v"(ga) : "memory");
    }
    const char* vg = (const char*)(vb + key0);        // 64 strided rows
    #pragma unroll
    for (int i = 0; i < 8; ++i) {
        int seg = i * 32 + lane;                      // 256 x 16B segments
        unsigned int       ld = (unsigned int)(size_t)((char*)vdst + (seg >> 2) * (VSTR * 2) + (seg & 3) * 16);
        unsigned long long ga = (unsigned long long)(size_t)(vg + (seg >> 2) * (SEQ * 2) + (seg & 3) * 16);
        asm volatile("global_load_async_to_lds_b128 %0, %1, off"
                     :: "v"(ld), "v"(ga) : "memory");
    }
}

static __device__ __forceinline__ v16h frag2(const _Float16* p) {
    v8h lo = *(const v8h*)p;
    v8h hi = *(const v8h*)(p + 8);
    v16h f;
    #pragma unroll
    for (int i = 0; i < 8; ++i) { f[i] = lo[i]; f[i + 8] = hi[i]; }
    return f;
}

__global__ __launch_bounds__(32) void attn_kernel(
    const _Float16* __restrict__ qf, const _Float16* __restrict__ kf,
    const _Float16* __restrict__ vtf, float* __restrict__ out)
{
    __shared__ __align__(64) _Float16 kbuf[2][32 * KSTR];
    __shared__ __align__(64) _Float16 vbuf[2][64 * VSTR];
    __shared__ __align__(64) _Float16 plds[16][PSTR];

    const int lane = threadIdx.x & 31;
    const int lr   = lane & 15;
    const int lh   = lane >> 4;

    const int gw    = blockIdx.x;              // 0..511 query tiles
    const int b     = gw >> 7;                 // batch
    const int qt    = gw & 127;                // query tile within batch
    const int qrow0 = qt * 16;

    const _Float16* qb = qf  + (size_t)(b * SEQ) * HD;
    const _Float16* kb = kf  + (size_t)(b * SEQ) * HD;
    const _Float16* vb = vtf + (size_t)b * HD * SEQ;

    const int nchunk = (qt >> 1) + 1;          // 32-key chunks up to diagonal

    // Prologue: start chunk 0 streaming into buffer 0.
    async_copy_chunk(kb, vb, 0, kbuf[0], vbuf[0], lane);

    // Q A-fragments for K(head) = 0..31 and 32..63 (overlaps with async DMA)
    v16h aq0, aq1;
    {
        const _Float16* qr = qb + (qrow0 + lr) * HD;
        aq0 = frag2(&qr[lh * 16]);     // low/high runs handled below
        v8h h0 = *(const v8h*)&qr[8 + lh * 16];
        v8h l1 = *(const v8h*)&qr[32 + lh * 16];
        v8h h1 = *(const v8h*)&qr[40 + lh * 16];
        #pragma unroll
        for (int i = 0; i < 8; ++i) {
            aq0[i + 8] = h0[i];
            aq1[i] = l1[i]; aq1[i + 8] = h1[i];
        }
    }

    v8f o0 = {}, o1 = {}, o2 = {}, o3 = {};
    float mrow[8], lrow[8];
    #pragma unroll
    for (int r = 0; r < 8; ++r) { mrow[r] = -1e30f; lrow[r] = 0.0f; }

    for (int ch = 0; ch < nchunk; ++ch) {
        const int key0 = ch * 32;

        // Kick off next chunk into the other buffer (clamped dummy on last
        // iteration keeps the ASYNCcnt invariant branch-free).
        const int nk = (ch + 1 < nchunk) ? key0 + 32 : key0;
        async_copy_chunk(kb, vb, nk, kbuf[(ch + 1) & 1], vbuf[(ch + 1) & 1], lane);

        // Outstanding = 32; waiting <=16 drains the in-order older batch
        // (this chunk's buffer) while next chunk keeps streaming.
        asm volatile("s_wait_asynccnt 0x10" ::: "memory");

        const _Float16* kt = kbuf[ch & 1];
        const _Float16* vt = vbuf[ch & 1];

        // ---- scores: S = Q(16x64) @ K^T(64x32) -> two 16x16 f32 tiles ----
        v8f s0 = {}, s1 = {};
        {
            v16h b0a = frag2(&kt[lr * KSTR + lh * 16]);
            v16h b0b = frag2(&kt[lr * KSTR + 32 + lh * 16]);
            v16h b1a = frag2(&kt[(16 + lr) * KSTR + lh * 16]);
            v16h b1b = frag2(&kt[(16 + lr) * KSTR + 32 + lh * 16]);
            s0 = wmma_f16(aq0, b0a, s0);
            s0 = wmma_f16(aq1, b0b, s0);
            s1 = wmma_f16(aq0, b1a, s1);
            s1 = wmma_f16(aq1, b1b, s1);
        }

        // ---- causal mask + scale into log2 domain, row max over 32 cols ----
        float rmax[8];
        #pragma unroll
        for (int r = 0; r < 8; ++r) {
            int rowp = qrow0 + r + 8 * lh;
            int c0 = key0 + lr, c1 = key0 + 16 + lr;
            float v0 = (c0 <= rowp) ? s0[r] * SCALE_LOG2 : -1e30f;
            float v1 = (c1 <= rowp) ? s1[r] * SCALE_LOG2 : -1e30f;
            s0[r] = v0; s1[r] = v1;
            rmax[r] = fmaxf(v0, v1);
        }
        #pragma unroll
        for (int r = 0; r < 8; ++r)
            #pragma unroll
            for (int mk = 1; mk < 16; mk <<= 1)
                rmax[r] = fmaxf(rmax[r], __shfl_xor(rmax[r], mk, 32));

        // ---- online softmax update (base-2) ----
        float corr[8], rsum[8];
        #pragma unroll
        for (int r = 0; r < 8; ++r) {
            float mn = fmaxf(mrow[r], rmax[r]);
            corr[r]  = exp2f(mrow[r] - mn);
            mrow[r]  = mn;
            float p0 = exp2f(s0[r] - mn);
            float p1 = exp2f(s1[r] - mn);
            s0[r] = p0; s1[r] = p1;
            rsum[r] = p0 + p1;
        }
        #pragma unroll
        for (int r = 0; r < 8; ++r)
            #pragma unroll
            for (int mk = 1; mk < 16; mk <<= 1)
                rsum[r] += __shfl_xor(rsum[r], mk, 32);
        #pragma unroll
        for (int r = 0; r < 8; ++r) {
            lrow[r] = lrow[r] * corr[r] + rsum[r];
            o0[r] *= corr[r]; o1[r] *= corr[r];
            o2[r] *= corr[r]; o3[r] *= corr[r];
        }

        // ---- transpose P (C layout -> A layout) through LDS ----
        #pragma unroll
        for (int r = 0; r < 8; ++r) {
            plds[r + 8 * lh][lr]      = (_Float16)s0[r];
            plds[r + 8 * lh][16 + lr] = (_Float16)s1[r];
        }
        // wave-synchronous LDS round trip: DS pipe is in-order per wave;
        // block compiler reordering and drain DScnt explicitly.
        asm volatile("s_wait_dscnt 0" ::: "memory");
        v16h ap;
        {
            v8h lo = *(const v8h*)&plds[lr][lh * 16];
            v8h hi = *(const v8h*)&plds[lr][lh * 16 + 8];
            #pragma unroll
            for (int i = 0; i < 8; ++i) { ap[i] = lo[i]; ap[i + 8] = hi[i]; }
        }

        // ---- O += P(16x32) @ V(32x64): 4 head tiles from LDS ----
        {
            v16h bv0 = frag2(&vt[(0  + lr) * VSTR + lh * 16]);
            v16h bv1 = frag2(&vt[(16 + lr) * VSTR + lh * 16]);
            v16h bv2 = frag2(&vt[(32 + lr) * VSTR + lh * 16]);
            v16h bv3 = frag2(&vt[(48 + lr) * VSTR + lh * 16]);
            o0 = wmma_f16(ap, bv0, o0);
            o1 = wmma_f16(ap, bv1, o1);
            o2 = wmma_f16(ap, bv2, o2);
            o3 = wmma_f16(ap, bv3, o3);
        }
    }

    // ---- epilogue: divide by softmax denominator, store fp32 ----
    // (trailing dummy async batch is drained by S_ENDPGM's implicit wait-idle)
    #pragma unroll
    for (int r = 0; r < 8; ++r) {
        float inv = 1.0f / lrow[r];
        int row = qrow0 + r + 8 * lh;
        float* op = out + (size_t)(b * SEQ + row) * HD;
        op[lr]      = o0[r] * inv;
        op[16 + lr] = o1[r] * inv;
        op[32 + lr] = o2[r] * inv;
        op[48 + lr] = o3[r] * inv;
    }
}

// ---------------------------------------------------------------------------
extern "C" void kernel_launch(void* const* d_in, const int* in_sizes, int n_in,
                              void* d_out, int out_size, void* d_ws, size_t ws_size,
                              hipStream_t stream)
{
    (void)in_sizes; (void)n_in; (void)out_size; (void)ws_size;
    const float* x  = (const float*)d_in[0];
    const float* Wq = (const float*)d_in[1];
    const float* bq = (const float*)d_in[2];
    const float* Wk = (const float*)d_in[3];
    const float* bk = (const float*)d_in[4];
    const float* Wv = (const float*)d_in[5];
    const float* bv = (const float*)d_in[6];

    char* ws = (char*)d_ws;
    _Float16* Wt  = (_Float16*)(ws);                         // 393,216 B
    _Float16* qf  = (_Float16*)(ws + 0x60000);               // 1 MB
    _Float16* kf  = (_Float16*)(ws + 0x60000 + 0x100000);    // 1 MB
    _Float16* vtf = (_Float16*)(ws + 0x60000 + 0x200000);    // 1 MB

    wt_kernel <<<768, 256, 0, stream>>>(Wq, Wk, Wv, Wt);
    qkv_kernel<<<ROWS / 16, 384, 0, stream>>>(x, Wt, bq, bk, bv, qf, kf, vtf);
    attn_kernel<<<BATCH * SEQ / 16, 32, 0, stream>>>(qf, kf, vtf, (float*)d_out);
}
